// GatedDeltaNet_36653250904139
// MI455X (gfx1250) — compile-verified, matrix operand order checked by reading, not run on hardware
//
#include <hip/hip_runtime.h>
#include <math.h>

#define BB 2
#define TT 2048
#define EE 2048
#define HH 16
#define DD 128
#define CC 2048
#define KCONV 4

typedef __attribute__((ext_vector_type(16))) __bf16 v16bf;
typedef __attribute__((ext_vector_type(8)))  float  v8f;
typedef __attribute__((ext_vector_type(8)))  unsigned short ush8;

union AFrag { v16bf v; ush8 h[2]; };

__device__ __forceinline__ float sigmoidf_(float x) {
  // 1/(1+e^-x) with hardware v_rcp_f32 instead of IEEE-div sequence
  return __builtin_amdgcn_rcpf(1.0f + __expf(-x));
}

// ---------------------------------------------------------------------------
// Tiled bf16 WMMA GEMM: Y[M,N] = X[M,Kd] @ W[N,Kd]^T, optional sigmoid.
// 128x128 tile per 256-thread block, 8 waves x (16x128 strip).
// Pipelined: double-buffered LDS (global prefetch under WMMA) and a
// depth-2 software pipeline on the B-fragment LDS loads.
// ---------------------------------------------------------------------------
#define LDSS 40   // padded row stride (halves) to break bank conflicts

__global__ __launch_bounds__(256) void gemm_bf16_wmma(
    const float* __restrict__ X, const float* __restrict__ W,
    float* __restrict__ Y, int M, int N, int Kd, int act)
{
  __shared__ __bf16 sA[2][128 * LDSS];
  __shared__ __bf16 sB[2][128 * LDSS];

  const int tid  = threadIdx.x;
  const int wave = tid >> 5;
  const int lane = tid & 31;
  const int m0   = blockIdx.y * 128;
  const int n0   = blockIdx.x * 128;

  v8f acc[8];
#pragma unroll
  for (int i = 0; i < 8; ++i) acc[i] = (v8f){0.f,0.f,0.f,0.f,0.f,0.f,0.f,0.f};

  const int mlane = lane & 15;
  const int hi    = lane >> 4;
  const int g0    = hi ? 8 : 0;     // K-group offsets per documented layout
  const int g1    = g0 + 16;
  const int arow  = wave * 16 + mlane;

  float4 xr[4], wr[4];

  auto gload = [&](int k0) {
#pragma unroll
    for (int i = 0; i < 4; ++i) {
      const int lin = tid + i * 256;        // 0..1023
      const int r   = lin >> 3;
      const int c4  = (lin & 7) << 2;
      xr[i] = *(const float4*)(X + (size_t)(m0 + r) * Kd + k0 + c4);
      wr[i] = *(const float4*)(W + (size_t)(n0 + r) * Kd + k0 + c4);
    }
  };
  auto lstore = [&](int sel) {
#pragma unroll
    for (int i = 0; i < 4; ++i) {
      const int lin = tid + i * 256;
      const int r   = lin >> 3;
      const int c4  = (lin & 7) << 2;
      __bf16* pa = &sA[sel][r * LDSS + c4];
      pa[0] = (__bf16)xr[i].x; pa[1] = (__bf16)xr[i].y;
      pa[2] = (__bf16)xr[i].z; pa[3] = (__bf16)xr[i].w;
      __bf16* pb = &sB[sel][r * LDSS + c4];
      pb[0] = (__bf16)wr[i].x; pb[1] = (__bf16)wr[i].y;
      pb[2] = (__bf16)wr[i].z; pb[3] = (__bf16)wr[i].w;
    }
  };

  const int nk = Kd >> 5;          // K-tiles of 32
  gload(0);
  lstore(0);

  for (int kt = 0; kt < nk; ++kt) {
    __syncthreads();
    const int sel = kt & 1;
    if (kt + 1 < nk) gload((kt + 1) << 5);   // prefetch next tile (global)

    const __bf16* pA = sA[sel];
    const __bf16* pB = sB[sel];

    AFrag a;
    a.h[0] = *(const ush8*)(pA + arow * LDSS + g0);
    a.h[1] = *(const ush8*)(pA + arow * LDSS + g1);

    // depth-2 pipeline on B fragments
    AFrag b0, b1;
    b0.h[0] = *(const ush8*)(pB + mlane * LDSS + g0);
    b0.h[1] = *(const ush8*)(pB + mlane * LDSS + g1);
    b1.h[0] = *(const ush8*)(pB + (16 + mlane) * LDSS + g0);
    b1.h[1] = *(const ush8*)(pB + (16 + mlane) * LDSS + g1);

#pragma unroll
    for (int nt = 0; nt < 8; ++nt) {
      AFrag bn;
      if (nt < 6) {
        const int brow = (nt + 2) * 16 + mlane;
        bn.h[0] = *(const ush8*)(pB + brow * LDSS + g0);
        bn.h[1] = *(const ush8*)(pB + brow * LDSS + g1);
      }
      acc[nt] = __builtin_amdgcn_wmma_f32_16x16x32_bf16(
          false, a.v, false, b0.v, (short)0, acc[nt], false, false);
      b0 = b1;
      if (nt < 6) b1 = bn;
    }

    if (kt + 1 < nk) lstore(sel ^ 1);        // fill other LDS buffer
  }

  // C/D layout: lane 0-15 -> N=lane, M=r; lane 16-31 -> N=lane-16, M=8+r.
  const int mb = m0 + wave * 16 + hi * 8;
#pragma unroll
  for (int nt = 0; nt < 8; ++nt) {
    const int col = n0 + nt * 16 + mlane;
#pragma unroll
    for (int r = 0; r < 8; ++r) {
      float val = acc[nt][r];
      if (act == 1) val = sigmoidf_(val);
      Y[(size_t)(mb + r) * N + col] = val;
    }
  }
}

// ---------------------------------------------------------------------------
// Causal depthwise conv (K=4) + SiLU + scale.
// ---------------------------------------------------------------------------
__global__ __launch_bounds__(256) void conv_silu(
    const float* __restrict__ U, const float* __restrict__ Wc,
    float* __restrict__ Yo, float scale)
{
  const int idx = blockIdx.x * 256 + threadIdx.x;   // B*T*C
  const int c = idx % CC;
  const int t = (idx / CC) % TT;
  const int b = idx / (CC * TT);
  float acc = 0.f;
#pragma unroll
  for (int j = 0; j < KCONV; ++j) {
    const int tt = t - (KCONV - 1) + j;
    if (tt >= 0)
      acc = fmaf(Wc[c * KCONV + j], U[((size_t)(b * TT + tt)) * CC + c], acc);
  }
  const float s = acc * sigmoidf_(acc);             // SiLU
  Yo[idx] = s * scale;
}

// ---------------------------------------------------------------------------
// beta = sigmoid(x @ Wbeta^T)   (B*T rows, 16 heads)
// ---------------------------------------------------------------------------
__global__ __launch_bounds__(256) void beta_proj(
    const float* __restrict__ X, const float* __restrict__ Wb,
    float* __restrict__ Beta)
{
  __shared__ float red[256];
  const int row  = blockIdx.x;          // b*T + t
  const int tid  = threadIdx.x;
  const int h    = tid & 15;
  const int part = tid >> 4;            // 16 parts of 128
  const float* xp = X + (size_t)row * EE + part * 128;
  const float* wp = Wb + (size_t)h * EE + part * 128;
  float acc = 0.f;
#pragma unroll 4
  for (int i = 0; i < 128; ++i) acc = fmaf(xp[i], wp[i], acc);
  red[tid] = acc;
  __syncthreads();
  if (tid < 16) {
    float s = 0.f;
#pragma unroll
    for (int p = 0; p < 16; ++p) s += red[p * 16 + tid];
    Beta[(size_t)row * HH + tid] = sigmoidf_(s);
  }
}

// ---------------------------------------------------------------------------
// Delta-rule scan. Grid (4 e-chunks, H, B); 256 threads; state in registers.
// Thread (el = tid>>3, dseg = tid&7) owns S[dseg*16 .. +15][e], e = ec*32+el:
// the 8 d-segment partials of a column live in 8 ADJACENT LANES of one wave,
// so the per-step reduction is 3 lane shuffles (no LDS, no barrier).
// Accumulates GroupNorm sum/sum^2 (merged via atomics) and writes the final
// recurrent state into SOut (second region of d_out).
// ---------------------------------------------------------------------------
__global__ __launch_bounds__(256) void delta_scan(
    const float* __restrict__ Q, const float* __restrict__ Kc,
    const float* __restrict__ V, const float* __restrict__ Beta,
    const float* __restrict__ S0, float* __restrict__ O,
    float* __restrict__ Stats, float* __restrict__ SOut)
{
  __shared__ float sq[128], sk[128], rstat[64];
  const int ec   = blockIdx.x;
  const int h    = blockIdx.y;
  const int b    = blockIdx.z;
  const int tid  = threadIdx.x;
  const int el   = tid >> 3;        // 0..31, e-local
  const int dseg = tid & 7;         // 0..7, d-segment (adjacent lanes)
  const int e    = ec * 32 + el;
  const int d0   = dseg * 16;

  float S[16];
  const size_t sbase = (((size_t)b * HH + h) * DD + d0) * DD + e;
#pragma unroll
  for (int i = 0; i < 16; ++i) S[i] = S0[sbase + (size_t)i * DD];

  float lsum = 0.f, lsq = 0.f;

  for (int t = 0; t < TT; ++t) {
    const size_t base = ((size_t)(b * TT + t) * HH + h) * DD;
    __syncthreads();
    if (tid < 128) sq[tid] = Q[base + tid];
    else           sk[tid - 128] = Kc[base + tid - 128];
    const float vv = V[base + e];
    const float bt = Beta[(size_t)(b * TT + t) * HH + h];
    __syncthreads();

    float part = 0.f;
#pragma unroll
    for (int i = 0; i < 16; ++i) {
      const float kd = sk[d0 + i];
      const float qd = sq[d0 + i];
      float s = S[i];
      s = fmaf(bt, fmaf(kd, vv, -s), s);   // S = (1-b)S + b*k*v
      S[i] = s;
      part = fmaf(s, qd, part);            // o[e] += S[d,e]*q[d]
    }
    // reduce the 8 d-segment partials within the wave (lanes xor 1,2,4)
    part += __shfl_xor(part, 1);
    part += __shfl_xor(part, 2);
    part += __shfl_xor(part, 4);
    if (dseg == 0) {
      O[base + e] = part;
      lsum += part;
      lsq = fmaf(part, part, lsq);
    }
  }

#pragma unroll
  for (int i = 0; i < 16; ++i) SOut[sbase + (size_t)i * DD] = S[i];

  if (dseg == 0) { rstat[el] = lsum; rstat[32 + el] = lsq; }
  __syncthreads();
  if (tid == 0) {
    float s = 0.f, q2 = 0.f;
    for (int j = 0; j < 32; ++j) { s += rstat[j]; q2 += rstat[32 + j]; }
    atomicAdd(&Stats[(b * HH + h) * 2],     s);
    atomicAdd(&Stats[(b * HH + h) * 2 + 1], q2);
  }
}

__global__ void zero_stats(float* p, int n) {
  int i = blockIdx.x * 64 + threadIdx.x;
  if (i < n) p[i] = 0.f;
}

// GroupNorm (per b,h over T*D) then multiply by gate.
__global__ __launch_bounds__(256) void norm_gate(
    const float* __restrict__ O, const float* __restrict__ Stats,
    const float* __restrict__ G, float* __restrict__ Out)
{
  const int idx = blockIdx.x * 256 + threadIdx.x;   // B*T*C
  const int cc  = idx % CC;
  const int b   = (idx / CC) / TT;
  const int h   = cc / DD;
  const float cnt  = (float)(TT * DD);
  const float mean = Stats[(b * HH + h) * 2] / cnt;
  const float var  = Stats[(b * HH + h) * 2 + 1] / cnt - mean * mean;
  const float inv  = rsqrtf(var + 1e-5f);
  Out[idx] = (O[idx] - mean) * inv * G[idx];
}

// ---------------------------------------------------------------------------
extern "C" void kernel_launch(void* const* d_in, const int* in_sizes, int n_in,
                              void* d_out, int out_size, void* d_ws, size_t ws_size,
                              hipStream_t stream) {
  const float* x      = (const float*)d_in[0];
  const float* rstate = (const float*)d_in[1];
  const float* Wq     = (const float*)d_in[2];
  const float* Wk     = (const float*)d_in[3];
  const float* Wv     = (const float*)d_in[4];
  const float* Wbeta  = (const float*)d_in[5];
  const float* Wgate  = (const float*)d_in[6];
  const float* Wo     = (const float*)d_in[7];
  const float* wqc    = (const float*)d_in[8];
  const float* wkc    = (const float*)d_in[9];
  const float* wvc    = (const float*)d_in[10];

  float* y         = (float*)d_out;
  float* state_out = y + (size_t)BB * TT * EE;

  float* ws = (float*)d_ws;
  const size_t NBT = (size_t)BB * TT;       // 4096
  const size_t SZ  = NBT * CC;              // 8388608 floats

  float* qpre  = ws;
  float* kpre  = ws + SZ;
  float* vpre  = ws + 2 * SZ;
  float* qpost = ws + 3 * SZ;
  float* kpost = ws + 4 * SZ;
  float* vpost = ws + 5 * SZ;
  float* gate  = ws + 6 * SZ;
  float* obuf  = ws + 7 * SZ;
  float* betab = ws + 8 * SZ;               // B*T*H
  float* stats = ws + 8 * SZ + NBT * HH;    // 2*B*H
  float* gated = qpre;                      // reuse (qpre dead after conv)

  const dim3 gg(CC / 128, (BB * TT) / 128); // (16, 32)
  const dim3 gb(256);
  const int  M = BB * TT;                   // 4096

  zero_stats<<<1, 64, 0, stream>>>(stats, 2 * BB * HH);

  gemm_bf16_wmma<<<gg, gb, 0, stream>>>(x, Wq,    qpre, M, CC, EE, 0);
  gemm_bf16_wmma<<<gg, gb, 0, stream>>>(x, Wk,    kpre, M, CC, EE, 0);
  gemm_bf16_wmma<<<gg, gb, 0, stream>>>(x, Wv,    vpre, M, CC, EE, 0);
  gemm_bf16_wmma<<<gg, gb, 0, stream>>>(x, Wgate, gate, M, CC, EE, 1);

  beta_proj<<<M, 256, 0, stream>>>(x, Wbeta, betab);

  conv_silu<<<(int)(SZ / 256), 256, 0, stream>>>(qpre, wqc, qpost,
                                                 0.08838834764831845f); // D^-0.5
  conv_silu<<<(int)(SZ / 256), 256, 0, stream>>>(kpre, wkc, kpost, 1.0f);
  conv_silu<<<(int)(SZ / 256), 256, 0, stream>>>(vpre, wvc, vpost, 1.0f);

  delta_scan<<<dim3(4, HH, BB), 256, 0, stream>>>(qpost, kpost, vpost, betab,
                                                  rstate, obuf, stats, state_out);

  norm_gate<<<(int)(SZ / 256), 256, 0, stream>>>(obuf, stats, gate, gated);

  gemm_bf16_wmma<<<gg, gb, 0, stream>>>(gated, Wo, y, M, EE, CC, 0);
}
